// RCNN_15487652069380
// MI455X (gfx1250) — compile-verified, matrix operand order checked by reading, not run on hardware
//
#include <hip/hip_runtime.h>
#include <hip/hip_bf16.h>
#include <math.h>

// ---------------------------------------------------------------------------
// CDNA5 (gfx1250) wave32 WMMA types & fragment helpers
// ---------------------------------------------------------------------------
typedef __bf16 bf16_t;
typedef __attribute__((ext_vector_type(16))) __bf16 v16bf;
typedef __attribute__((ext_vector_type(8)))  __bf16 v8bf;
typedef __attribute__((ext_vector_type(8)))  float  v8f;
typedef __attribute__((ext_vector_type(4)))  float  f32x4;

union CFrag { v8f v; float e[8]; };

// K index inside a 16x32 bf16 A/B fragment for element e, half 'hi'
// (ISA 7.12.2: lanes0-15 hold K=0..7,16..23; lanes16-31 hold K=8..15,24..31)
__device__ __forceinline__ int frag_k(int e, int hi) {
  return (e < 8) ? (hi * 8 + e) : (8 + hi * 8 + e);
}
// inverse map: k (0..31) -> lane-half hi and element e
__device__ __forceinline__ void frag_inv(int k, int& hi, int& e) {
  hi = (k >> 3) & 1;
  e  = (k < 16) ? (k - hi * 8) : (k - 8 - hi * 8);
}

__device__ __forceinline__ v8bf pack8(f32x4 a, f32x4 b) {
  v8bf r;
  r[0] = (bf16_t)a[0]; r[1] = (bf16_t)a[1]; r[2] = (bf16_t)a[2]; r[3] = (bf16_t)a[3];
  r[4] = (bf16_t)b[0]; r[5] = (bf16_t)b[1]; r[6] = (bf16_t)b[2]; r[7] = (bf16_t)b[3];
  return r;
}

// ---------------------------------------------------------------------------
// Tensor Data Mover: 1-D contiguous f32 tile, global -> LDS (ISA ch.8 D#)
// ---------------------------------------------------------------------------
typedef __attribute__((ext_vector_type(4))) unsigned int tdm_ui4;
typedef __attribute__((ext_vector_type(8))) int tdm_i8;
typedef __attribute__((ext_vector_type(4))) int tdm_i4;

__device__ __forceinline__ void tdm_load_f32_1d(void* ldsDst, const void* gsrc,
                                                int nElems) {
  unsigned long long ga = (unsigned long long)(size_t)gsrc;
  unsigned int lds = (unsigned int)(size_t)ldsDst;
  tdm_ui4 g0;
  g0[0] = 1u;                                   // count=1 user descriptor
  g0[1] = lds;                                  // lds_addr (bytes)
  g0[2] = (unsigned int)ga;                     // global_addr[31:0]
  g0[3] = (unsigned int)((ga >> 32) & 0x1FFFFFFull) | 0x80000000u; // hi + type=2
  tdm_i8 g1;
  g1[0] = (int)(2u << 16);                      // data_size = 4 bytes
  g1[1] = (int)(((unsigned)nElems & 0xFFFFu) << 16);  // tensor_dim0[15:0]
  g1[2] = (int)((unsigned)nElems >> 16);              // tensor_dim0[31:16]
  g1[3] = (int)(((unsigned)nElems & 0xFFFFu) << 16);  // tile_dim0
  g1[4] = 0;
  g1[5] = nElems;                               // tensor_dim0_stride lo
  g1[6] = 0;
  g1[7] = 0;
  tdm_i4 z4 = {0, 0, 0, 0};
#if defined(__clang_major__) && __clang_major__ >= 23
  tdm_i8 z8 = {0, 0, 0, 0, 0, 0, 0, 0};
  __builtin_amdgcn_tensor_load_to_lds(g0, g1, z4, z4, z8, 0);
#else
  __builtin_amdgcn_tensor_load_to_lds(g0, g1, z4, z4, 0);
#endif
}

// ---------------------------------------------------------------------------
// Staging helpers: write a 64x32 source block into fragment-order LDS.
// "nt" = source rows are the 64-dim, contiguous along k (one 16B LDS store).
// "t"  = source rows are the 32 k-dim, contiguous along the 64-dim (scatter).
// K must be a multiple of 32 (no guards).
// ---------------------------------------------------------------------------
__device__ __forceinline__ void stage_nt(const float* __restrict__ base, int ld,
                                         int r0, int k0,
                                         bf16_t (*lds)[32][16], int tid) {
  int r = tid >> 2, kb = (tid & 3) * 8;
  const float* src = base + (size_t)(r0 + r) * ld + (k0 + kb);
  f32x4 a = *(const f32x4*)src;
  f32x4 b = *(const f32x4*)(src + 4);
  int hh = (kb >> 3) & 1;
  int e0 = (kb & 16) ? 8 : 0;
  *(v8bf*)&lds[r >> 4][hh * 16 + (r & 15)][e0] = pack8(a, b);
}

__device__ __forceinline__ void stage_t(const float* __restrict__ base, int ld,
                                        int r0, int k0,
                                        bf16_t (*lds)[32][16], int tid) {
  int k = tid >> 3, rb = (tid & 7) * 8;
  const float* src = base + (size_t)(k0 + k) * ld + (r0 + rb);
  f32x4 a = *(const f32x4*)src;
  f32x4 b = *(const f32x4*)(src + 4);
  int hh, e; frag_inv(k, hh, e);
  float v[8] = {a[0], a[1], a[2], a[3], b[0], b[1], b[2], b[3]};
#pragma unroll
  for (int e8 = 0; e8 < 8; ++e8) {
    int r = rb + e8;
    lds[r >> 4][hh * 16 + (r & 15)][e] = (bf16_t)v[e8];
  }
}

// ---------------------------------------------------------------------------
// Tiled GEMM: 64x64 C tile / 256-thread block, double-buffered LDS,
// one barrier per k-step. Requires M%64==0, N%64==0, K%32==0.
// ---------------------------------------------------------------------------
__global__ __launch_bounds__(256)
void gemm_wmma_tiled_kernel(const float* __restrict__ A, const float* __restrict__ B,
                            const float* __restrict__ bias, float* __restrict__ C,
                            int K, int lda, int ldb, int ldc,
                            int transA, int transB, int relu) {
  __shared__ __align__(32) bf16_t aLds[2][4][32][16];
  __shared__ __align__(32) bf16_t bLds[2][4][32][16];
  int m0 = blockIdx.y * 64;
  int n0 = blockIdx.x * 64;
  int tid = threadIdx.x;
  int wave = tid >> 5, lane = tid & 31;
  int mt = wave & 3, ng = wave >> 2;

  CFrag acc0, acc1;
#pragma unroll
  for (int r = 0; r < 8; ++r) { acc0.e[r] = 0.f; acc1.e[r] = 0.f; }

  auto stageA = [&](int k0, int buf) {
    if (!transA) stage_nt(A, lda, m0, k0, aLds[buf], tid);
    else         stage_t (A, lda, m0, k0, aLds[buf], tid);
  };
  auto stageB = [&](int k0, int buf) {
    if (!transB) stage_t (B, ldb, n0, k0, bLds[buf], tid);
    else         stage_nt(B, ldb, n0, k0, bLds[buf], tid);
  };

  stageA(0, 0);
  stageB(0, 0);
  __syncthreads();
  int nk = K >> 5;
  for (int i = 0; i < nk; ++i) {
    int cur = i & 1, nxt = cur ^ 1;
    if (i + 1 < nk) { stageA((i + 1) << 5, nxt); stageB((i + 1) << 5, nxt); }
    v16bf a  = *(const v16bf*)&aLds[cur][mt][lane][0];
    v16bf b0 = *(const v16bf*)&bLds[cur][ng * 2 + 0][lane][0];
    v16bf b1 = *(const v16bf*)&bLds[cur][ng * 2 + 1][lane][0];
    acc0.v = __builtin_amdgcn_wmma_f32_16x16x32_bf16(
        false, a, false, b0, (short)0, acc0.v, false, false);
    acc1.v = __builtin_amdgcn_wmma_f32_16x16x32_bf16(
        false, a, false, b1, (short)0, acc1.v, false, false);
    __syncthreads();
  }

  int nlo = lane & 15, hi = lane >> 4;
  int c0 = n0 + (ng * 2) * 16 + nlo;
  int c1 = c0 + 16;
  float bv0 = bias ? bias[c0] : 0.f;
  float bv1 = bias ? bias[c1] : 0.f;
#pragma unroll
  for (int r = 0; r < 8; ++r) {
    int m = m0 + mt * 16 + r + 8 * hi;
    float v0 = acc0.e[r] + bv0;
    float v1 = acc1.e[r] + bv1;
    if (relu) { v0 = fmaxf(v0, 0.f); v1 = fmaxf(v1, 0.f); }
    C[(size_t)m * ldc + c0] = v0;
    C[(size_t)m * ldc + c1] = v1;
  }
}

// ---------------------------------------------------------------------------
// Pack (Nrows x Kcols) row-major weight into B-fragment-order bf16 blob,
// zero-padding rows up to Npad (multiple of 16). Kcols must be %32.
// ---------------------------------------------------------------------------
__global__ void pack_b_kernel(const float* __restrict__ W, bf16_t* __restrict__ blob,
                              int Nrows, int Npad, int Kcols) {
  int idx = blockIdx.x * blockDim.x + threadIdx.x;
  int total = Npad * Kcols;
  if (idx >= total) return;
  int e = idx & 15;
  int lane = (idx >> 4) & 31;
  int tile = idx >> 9;
  int KB = Kcols >> 5;
  int nt = tile / KB, kb = tile - nt * KB;
  int n = nt * 16 + (lane & 15);
  int k = kb * 32 + frag_k(e, (lane >> 4) & 1);
  float w = (n < Nrows) ? W[(size_t)n * Kcols + k] : 0.f;
  blob[idx] = (bf16_t)w;
}

// zero-pad Wih0 (384x20) -> (384x32)
__global__ void pad_wih0_kernel(const float* __restrict__ W, float* __restrict__ Wp) {
  int idx = blockIdx.x * blockDim.x + threadIdx.x;
  if (idx >= 384 * 32) return;
  int n = idx >> 5, k = idx & 31;
  Wp[idx] = (k < 20) ? W[n * 20 + k] : 0.f;
}

// ---------------------------------------------------------------------------
// Row-major A x packed-B(16 cols): 128 rows per block. M%128==0, K%32==0.
// ---------------------------------------------------------------------------
__global__ __launch_bounds__(256)
void gemm_rowA_packB16_kernel(const float* __restrict__ A,
                              const bf16_t* __restrict__ Bpack,
                              float* __restrict__ C,
                              int K, int lda, int ldc, int Nreal) {
  __shared__ __align__(32) bf16_t aLds[8][32][16];
  int m0 = blockIdx.x * 128;
  int tid = threadIdx.x, wave = tid >> 5, lane = tid & 31;
  CFrag acc;
#pragma unroll
  for (int r = 0; r < 8; ++r) acc.e[r] = 0.f;
  int KB = K >> 5;
  for (int kb = 0; kb < KB; ++kb) {
    int m = tid >> 1;
    int kh = (tid & 1) * 16;
    const float* src = A + (size_t)(m0 + m) * lda + kb * 32 + kh;
    f32x4 x0 = *(const f32x4*)src;
    f32x4 x1 = *(const f32x4*)(src + 4);
    f32x4 x2 = *(const f32x4*)(src + 8);
    f32x4 x3 = *(const f32x4*)(src + 12);
    int mtl = m >> 4, ml = m & 15;
    int e0 = kh ? 8 : 0;
    *(v8bf*)&aLds[mtl][ml][e0]      = pack8(x0, x1);  // hi = 0 half
    *(v8bf*)&aLds[mtl][16 + ml][e0] = pack8(x2, x3);  // hi = 1 half
    __syncthreads();
    v16bf a = *(const v16bf*)&aLds[wave][lane][0];
    v16bf b = *(const v16bf*)(Bpack + ((size_t)kb * 32 + lane) * 16);
    acc.v = __builtin_amdgcn_wmma_f32_16x16x32_bf16(
        false, a, false, b, (short)0, acc.v, false, false);
    __syncthreads();
  }
  int n = lane & 15, hi = lane >> 4;
  if (n < Nreal) {
#pragma unroll
    for (int r = 0; r < 8; ++r) {
      int m = m0 + wave * 16 + r + 8 * hi;
      C[(size_t)m * ldc + n] = acc.e[r];
    }
  }
}

// ---------------------------------------------------------------------------
// Conv branch: TDM async load of input slice into LDS, then 4x(conv+pool).
// ---------------------------------------------------------------------------
__device__ void conv_pool(float* bufIn, float* bufOut,
                          const float* __restrict__ w, const float* __restrict__ b,
                          int Cin, int Cout, int K, int& L) {
  int Lc = L - K + 1;
  for (int i = threadIdx.x; i < Cout * Lc; i += blockDim.x) {
    int co = i / Lc, l = i - co * Lc;
    float acc = b[co];
    const float* wr = w + (size_t)co * Cin * K;
    for (int ci = 0; ci < Cin; ++ci) {
      const float* xr = bufIn + ci * L + l;
      const float* wc = wr + ci * K;
      for (int k = 0; k < K; ++k) acc = fmaf(wc[k], xr[k], acc);
    }
    bufOut[i] = acc;
  }
  __syncthreads();
  int Lp = Lc >> 1;
  for (int i = threadIdx.x; i < Cout * Lp; i += blockDim.x) {
    int co = i / Lp, l = i - co * Lp;
    bufIn[i] = fmaxf(bufOut[co * Lc + 2 * l], bufOut[co * Lc + 2 * l + 1]);
  }
  __syncthreads();
  L = Lp;
}

__global__ __launch_bounds__(256)
void conv_branch_kernel(const float* __restrict__ x, int fOff, int Cin0,
                        const float* __restrict__ w0, const float* __restrict__ b0,
                        const float* __restrict__ w1, const float* __restrict__ b1,
                        const float* __restrict__ w2, const float* __restrict__ b2,
                        const float* __restrict__ w3, const float* __restrict__ b3,
                        int C1, int C2, int C3, int C4,
                        int K1, int K2, int K3, int K4,
                        float* __restrict__ fout, int Dout, int ldf) {
  __shared__ float bufA[4096];
  __shared__ float bufB[4096];
  int n = blockIdx.x;
  const float* xs = x + ((size_t)n * 49 + fOff) * 128;
  if (threadIdx.x < 32)
    tdm_load_f32_1d(bufA, xs, Cin0 * 128);
  __builtin_prefetch(w1, 0, 0);
  __builtin_prefetch(w2, 0, 0);
  __builtin_prefetch(w3, 0, 0);
  __builtin_amdgcn_s_wait_tensorcnt(0);
  __syncthreads();

  int L = 128;
  conv_pool(bufA, bufB, w0, b0, Cin0, C1, K1, L);
  conv_pool(bufA, bufB, w1, b1, C1, C2, K2, L);
  conv_pool(bufA, bufB, w2, b2, C2, C3, K3, L);
  conv_pool(bufA, bufB, w3, b3, C3, C4, K4, L);

  for (int i = threadIdx.x; i < Dout; i += 256)
    fout[(size_t)n * ldf + i] = bufA[i];
}

// ---------------------------------------------------------------------------
// PCA helpers
// ---------------------------------------------------------------------------
__global__ __launch_bounds__(256)
void colmean_kernel(const float* __restrict__ F, float* __restrict__ mean,
                    int N, int D) {
  __shared__ float sred[256];
  int d = blockIdx.x;
  float acc = 0.f;
  for (int i = threadIdx.x; i < N; i += 256) acc += F[(size_t)i * D + d];
  sred[threadIdx.x] = acc;
  __syncthreads();
  for (int s = 128; s > 0; s >>= 1) {
    if (threadIdx.x < s) sred[threadIdx.x] += sred[threadIdx.x + s];
    __syncthreads();
  }
  if (threadIdx.x == 0) mean[d] = sred[0] / (float)N;
}

__global__ void center_kernel(float* __restrict__ F, const float* __restrict__ mean,
                              int N, int D) {
  size_t total = (size_t)N * D;
  for (size_t i = (size_t)blockIdx.x * blockDim.x + threadIdx.x; i < total;
       i += (size_t)gridDim.x * blockDim.x)
    F[i] -= mean[i % D];
}

__global__ __launch_bounds__(1024)
void jacobi_topk_kernel(const float* __restrict__ cov, float* __restrict__ Aw,
                        float* __restrict__ Vw, float* __restrict__ Vk,
                        int D, int kk, int sweeps) {
  for (int i = threadIdx.x; i < D * D; i += blockDim.x) {
    Aw[i] = cov[i];
    Vw[i] = ((i / D) == (i % D)) ? 1.f : 0.f;
  }
  __syncthreads();
  __shared__ float cs[2];
  for (int sw = 0; sw < sweeps; ++sw) {
    for (int p = 0; p < D - 1; ++p) {
      for (int q = p + 1; q < D; ++q) {
        if (threadIdx.x == 0) {
          float app = Aw[(size_t)p * D + p];
          float aqq = Aw[(size_t)q * D + q];
          float apq = Aw[(size_t)p * D + q];
          float c = 1.f, s = 0.f;
          if (fabsf(apq) > 1e-12f * (fabsf(app) + fabsf(aqq)) + 1e-30f) {
            float th = (aqq - app) / (2.f * apq);
            float t = (th >= 0.f ? 1.f : -1.f) / (fabsf(th) + sqrtf(1.f + th * th));
            c = 1.f / sqrtf(1.f + t * t);
            s = t * c;
          }
          cs[0] = c; cs[1] = s;
        }
        __syncthreads();
        float c = cs[0], s = cs[1];
        if (!(c == 1.f && s == 0.f)) {
          for (int i = threadIdx.x; i < D; i += blockDim.x) {
            float aip = Aw[(size_t)i * D + p], aiq = Aw[(size_t)i * D + q];
            Aw[(size_t)i * D + p] = c * aip - s * aiq;
            Aw[(size_t)i * D + q] = s * aip + c * aiq;
            float vip = Vw[(size_t)i * D + p], viq = Vw[(size_t)i * D + q];
            Vw[(size_t)i * D + p] = c * vip - s * viq;
            Vw[(size_t)i * D + q] = s * vip + c * viq;
          }
          __syncthreads();
          for (int i = threadIdx.x; i < D; i += blockDim.x) {
            float api = Aw[(size_t)p * D + i], aqi = Aw[(size_t)q * D + i];
            Aw[(size_t)p * D + i] = c * api - s * aqi;
            Aw[(size_t)q * D + i] = s * api + c * aqi;
          }
        }
        __syncthreads();
      }
    }
  }
  __shared__ int sel[16];
  if (threadIdx.x == 0) {
    unsigned used[16];
    for (int i = 0; i < 16; ++i) used[i] = 0u;
    for (int j = 0; j < kk; ++j) {
      float best = -3.4e38f; int bi = 0;
      for (int d = 0; d < D; ++d) {
        if (used[d >> 5] & (1u << (d & 31))) continue;
        float ev = Aw[(size_t)d * D + d];
        if (ev > best) { best = ev; bi = d; }
      }
      used[bi >> 5] |= 1u << (bi & 31);
      sel[j] = bi;
    }
  }
  __syncthreads();
  for (int j = 0; j < kk; ++j) {
    int idx = sel[j];
    for (int d = threadIdx.x; d < D; d += blockDim.x)
      Vk[(size_t)j * D + d] = Vw[(size_t)d * D + idx];
  }
}

__global__ __launch_bounds__(256)
void signfix_kernel(float* __restrict__ X, int N, int ld, int col0) {
  int col = col0 + blockIdx.x;
  __shared__ float sv[256];
  __shared__ int   si[256];
  __shared__ float sgn;
  float best = -1.f; int bi = 0;
  for (int i = threadIdx.x; i < N; i += 256) {
    float v = fabsf(X[(size_t)i * ld + col]);
    if (v > best) { best = v; bi = i; }
  }
  sv[threadIdx.x] = best; si[threadIdx.x] = bi;
  __syncthreads();
  for (int s = 128; s > 0; s >>= 1) {
    if (threadIdx.x < s && sv[threadIdx.x + s] > sv[threadIdx.x]) {
      sv[threadIdx.x] = sv[threadIdx.x + s];
      si[threadIdx.x] = si[threadIdx.x + s];
    }
    __syncthreads();
  }
  if (threadIdx.x == 0)
    sgn = (X[(size_t)si[0] * ld + col] < 0.f) ? -1.f : 1.f;
  __syncthreads();
  float s = sgn;
  if (s != 1.f)
    for (int i = threadIdx.x; i < N; i += 256)
      X[(size_t)i * ld + col] *= s;
}

// ---------------------------------------------------------------------------
// Fused GRU step with pre-packed Whh (B-fragment order).
// ---------------------------------------------------------------------------
__global__ __launch_bounds__(256)
void gru_step_kernel(const float* __restrict__ Gi, int giRowStride,
                     const float* __restrict__ hin,
                     const bf16_t* __restrict__ WhhPack,  // [24][4][32][16]
                     const float* __restrict__ bhh,
                     float* __restrict__ hout,
                     float* __restrict__ ys, int ysRowStride) {
  const int H = 128, G3 = 384;
  __shared__ __align__(32) bf16_t aLds[4][32][16];
  __shared__ float sGh[16 * 384];
  int m0 = blockIdx.x * 16;
  int tid = threadIdx.x, wave = tid >> 5, lane = tid & 31;

  {  // stage hin tile (16 x 128) in fragment order: one 16B LDS store
    int m = tid >> 4;
    int kb = (tid & 15) * 8;
    const float* src = hin + (size_t)(m0 + m) * H + kb;
    f32x4 a = *(const f32x4*)src;
    f32x4 b = *(const f32x4*)(src + 4);
    int ks = kb >> 5, kk = kb & 31;
    int hh = (kk >> 3) & 1;
    int e0 = (kk & 16) ? 8 : 0;
    *(v8bf*)&aLds[ks][hh * 16 + (m & 15)][e0] = pack8(a, b);
  }
  __syncthreads();

  for (int tn = wave; tn < 24; tn += 8) {
    CFrag acc;
#pragma unroll
    for (int r = 0; r < 8; ++r) acc.e[r] = 0.f;
    const v16bf* bp = (const v16bf*)(WhhPack + (size_t)tn * 4 * 32 * 16);
#pragma unroll
    for (int kb = 0; kb < 4; ++kb) {
      v16bf a = *(const v16bf*)&aLds[kb][lane][0];
      v16bf b = bp[kb * 32 + lane];
      acc.v = __builtin_amdgcn_wmma_f32_16x16x32_bf16(
          false, a, false, b, (short)0, acc.v, false, false);
    }
    int n = tn * 16 + (lane & 15), hi = lane >> 4;
#pragma unroll
    for (int r = 0; r < 8; ++r)
      sGh[(r + 8 * hi) * G3 + n] = acc.e[r] + bhh[n];
  }
  __syncthreads();

  for (int i = tid; i < 16 * H; i += 256) {
    int mm = i / H, j = i - mm * H;
    int row = m0 + mm;
    const float* gi = Gi + (size_t)row * giRowStride;
    float ir = gi[j], iz = gi[H + j], in = gi[2 * H + j];
    float hr = sGh[mm * G3 + j];
    float hz = sGh[mm * G3 + H + j];
    float hn = sGh[mm * G3 + 2 * H + j];
    float r = 1.f / (1.f + expf(-(ir + hr)));
    float z = 1.f / (1.f + expf(-(iz + hz)));
    float nn = tanhf(in + r * hn);
    float hprev = hin[(size_t)row * H + j];
    float hv = (1.f - z) * nn + z * hprev;
    hout[(size_t)row * H + j] = hv;
    ys[(size_t)row * ysRowStride + j] = hv;
  }
}

// final head: y[m] = dot(T[m,:], w) + b
__global__ __launch_bounds__(256)
void head_dot_kernel(const float* __restrict__ Tm, const float* __restrict__ w,
                     const float* __restrict__ b, float* __restrict__ out,
                     int M, int K, int ldo) {
  int m = blockIdx.x * blockDim.x + threadIdx.x;
  if (m >= M) return;
  float acc = b[0];
  const float* row = Tm + (size_t)m * K;
  for (int k = 0; k < K; ++k) acc = fmaf(row[k], w[k], acc);
  out[(size_t)m * ldo] = acc;
}

__global__ void zero_kernel(float* __restrict__ p, size_t n) {
  size_t i = (size_t)blockIdx.x * blockDim.x + threadIdx.x;
  if (i < n) p[i] = 0.f;
}

// ---------------------------------------------------------------------------
// Host orchestration
// ---------------------------------------------------------------------------
extern "C" void kernel_launch(void* const* d_in, const int* in_sizes, int n_in,
                              void* d_out, int out_size, void* d_ws, size_t ws_size,
                              hipStream_t stream) {
  (void)in_sizes; (void)n_in; (void)out_size;
  const float* x = (const float*)d_in[0];
  const float* p1[8]; const float* p2[8]; const float* p3[8];
  for (int i = 0; i < 8; ++i) {
    p1[i] = (const float*)d_in[1 + i];
    p2[i] = (const float*)d_in[9 + i];
    p3[i] = (const float*)d_in[17 + i];
  }
  const float* wih0 = (const float*)d_in[25];
  const float* whh0 = (const float*)d_in[26];
  const float* bih0 = (const float*)d_in[27];
  const float* bhh0 = (const float*)d_in[28];
  const float* gWih = (const float*)d_in[29];
  const float* gWhh = (const float*)d_in[30];
  const float* gBih = (const float*)d_in[31];
  const float* gBhh = (const float*)d_in[32];
  const float* W11 = (const float*)d_in[33]; const float* b11 = (const float*)d_in[34];
  const float* W21 = (const float*)d_in[35]; const float* b21 = (const float*)d_in[36];
  const float* W12 = (const float*)d_in[37]; const float* b12 = (const float*)d_in[38];
  const float* W22 = (const float*)d_in[39]; const float* b22 = (const float*)d_in[40];
  float* out = (float*)d_out;

  const int Nsamp = 16384, Bt = 256, T = 64, H = 128, G3 = 384;

  float* ws = (float*)d_ws;
  size_t off = 0;
  auto alloc = [&](size_t n) { float* p = ws + off; off += n; return p; };
  float* f1   = alloc((size_t)Nsamp * 320);
  float* f2   = alloc((size_t)Nsamp * 128);   // 120 real + 8 zero pad
  float* f3   = alloc((size_t)Nsamp * 320);
  float* mean = alloc(320);
  float* cov  = alloc(320 * 320);
  float* Aw   = alloc(320 * 320);
  float* Vw   = alloc(320 * 320);
  float* Vk   = alloc(8 * 320);
  float* X    = alloc((size_t)Nsamp * 32);    // 20 real + 12 zero pad
  float* Gi   = alloc((size_t)Nsamp * G3);
  float* ys0  = alloc((size_t)Nsamp * H);
  float* ys1  = alloc((size_t)Nsamp * H);
  float* hA   = alloc((size_t)Bt * H);
  float* hB   = alloc((size_t)Bt * H);
  float* t1   = alloc((size_t)Bt * 256);
  float* wih0p = alloc((size_t)G3 * 32);      // zero-padded layer-0 Wih
  bf16_t* whhPack = (bf16_t*)alloc((size_t)G3 * H / 2);
  bf16_t* vkPack  = (bf16_t*)alloc((size_t)16 * 320 / 2);
  if (off * sizeof(float) > ws_size) return;

  zero_kernel<<<(int)(((size_t)Nsamp * 128 + 255) / 256), 256, 0, stream>>>(f2, (size_t)Nsamp * 128);
  zero_kernel<<<(int)(((size_t)Nsamp * 32 + 255) / 256), 256, 0, stream>>>(X, (size_t)Nsamp * 32);
  pad_wih0_kernel<<<(G3 * 32 + 255) / 256, 256, 0, stream>>>(wih0, wih0p);

  // ---- three conv branches + PCA ----
  struct Br {
    int fOff, Cin0, C1, C2, C3, C4, K1, K2, K3, K4, Dreal, Dpad, k, colOff;
    const float* const* p; float* f;
  };
  Br brs[3] = {
    { 0, 24, 32, 32, 64, 64, 5, 4, 3, 3, 320, 320, 8,  0, p1, f1},
    {24,  1, 12, 12, 24, 24, 5, 4, 3, 3, 120, 128, 4,  8, p2, f2},
    {25, 24, 32, 32, 64, 64, 5, 4, 3, 3, 320, 320, 8, 12, p3, f3},
  };
  for (int b = 0; b < 3; ++b) {
    Br& r = brs[b];
    conv_branch_kernel<<<Nsamp, 256, 0, stream>>>(
        x, r.fOff, r.Cin0,
        r.p[0], r.p[1], r.p[2], r.p[3], r.p[4], r.p[5], r.p[6], r.p[7],
        r.C1, r.C2, r.C3, r.C4, r.K1, r.K2, r.K3, r.K4, r.f, r.Dreal, r.Dpad);
    colmean_kernel<<<r.Dpad, 256, 0, stream>>>(r.f, mean, Nsamp, r.Dpad);
    center_kernel<<<512, 256, 0, stream>>>(r.f, mean, Nsamp, r.Dpad);
    dim3 gc(r.Dpad / 64, r.Dpad / 64);
    gemm_wmma_tiled_kernel<<<gc, 256, 0, stream>>>(r.f, r.f, nullptr, cov,
        Nsamp, r.Dpad, r.Dpad, r.Dpad, 1, 0, 0);
    jacobi_topk_kernel<<<1, 1024, 0, stream>>>(cov, Aw, Vw, Vk, r.Dpad, r.k, 4);
    pack_b_kernel<<<(16 * r.Dpad + 255) / 256, 256, 0, stream>>>(
        Vk, vkPack, r.k, 16, r.Dpad);
    gemm_rowA_packB16_kernel<<<Nsamp / 128, 256, 0, stream>>>(
        r.f, vkPack, X + r.colOff, r.Dpad, r.Dpad, 32, r.k);
    signfix_kernel<<<r.k, 256, 0, stream>>>(X, Nsamp, 32, r.colOff);
  }

  // ---- GRU stack ----
  auto run_gru = [&](const float* inSeq, int inDim, int inLd,
                     const float* wih, const float* whh,
                     const float* bih, const float* bhh,
                     float* ysOut, float*& hcur, float*& hnxt) {
    dim3 gg(G3 / 64, Nsamp / 64);
    gemm_wmma_tiled_kernel<<<gg, 256, 0, stream>>>(inSeq, wih, bih, Gi,
        inDim, inLd, inDim, G3, 0, 1, 0);
    pack_b_kernel<<<(G3 * H + 255) / 256, 256, 0, stream>>>(whh, whhPack, G3, G3, H);
    zero_kernel<<<(Bt * H + 255) / 256, 256, 0, stream>>>(hcur, (size_t)Bt * H);
    for (int t = 0; t < T; ++t) {
      gru_step_kernel<<<Bt / 16, 256, 0, stream>>>(
          Gi + (size_t)t * G3, T * G3, hcur, whhPack, bhh, hnxt,
          ysOut + (size_t)t * H, T * H);
      float* tmp = hcur; hcur = hnxt; hnxt = tmp;
    }
  };

  float* hcur = hA; float* hnxt = hB;
  run_gru(X, 32, 32, wih0p, whh0, bih0, bhh0, ys0, hcur, hnxt);
  float* ysIn = ys0; float* ysOut = ys1;
  for (int l = 0; l < 4; ++l) {
    run_gru(ysIn, H, H, gWih + (size_t)l * G3 * H, gWhh + (size_t)l * G3 * H,
            gBih + (size_t)l * G3, gBhh + (size_t)l * G3, ysOut, hcur, hnxt);
    float* tmp = ysIn; ysIn = ysOut; ysOut = tmp;
  }
  float* hlast = hcur;

  // ---- FC heads ----
  dim3 gfc(4, 4);
  gemm_wmma_tiled_kernel<<<gfc, 256, 0, stream>>>(hlast, W11, b11, t1,
      H, H, H, 256, 0, 1, 1);
  head_dot_kernel<<<1, 256, 0, stream>>>(t1, W21, b21, out, Bt, 256, 2);
  gemm_wmma_tiled_kernel<<<gfc, 256, 0, stream>>>(hlast, W12, b12, t1,
      H, H, H, 256, 0, 1, 1);
  head_dot_kernel<<<1, 256, 0, stream>>>(t1, W22, b22, out + 1, Bt, 256, 2);
}
